// LSTMTaggerWithCharEmbeds_13950053778304
// MI455X (gfx1250) — matrix operand order repaired and checked by reading, not compiled, not run on hardware
//
#include <hip/hip_runtime.h>

// ---------------------------------------------------------------------------
// LSTM tagger with char embeddings for gfx1250 (MI455X, wave32, WMMA + TDM).
//   S=8192 words, L=16 chars, WE=256, CE=64, CD=128, HD=512, TAGS=48
// Pipeline:
//   prep:   fp32 -> bf16 (transposed for B-fragment loads), bias sums
//   char:   16 words/WG, 16-step LSTM recurrence via v_wmma_f32_16x16x32_bf16;
//           weights staged to LDS once by the Tensor Data Mover (padded rows)
//   gemm:   Gx = X @ wW_ih^T (WMMA; X tile staged via global_load_async_to_lds)
//   rec:    sequential 8192-step word LSTM (latency-bound, single WG, VALU)
//   logits: outs @ W_out^T + b, fused log_softmax (WMMA + async LDS staging)
// ---------------------------------------------------------------------------

#define NW 8192      // words (S)
#define LC 16        // chars per word (L)
#define WE 256
#define CE 64
#define CD 128
#define HD 512
#define TAGS 48
#define XDIM (WE + CD)          // 384
#define CG (4 * CD)             // 512 char gates
#define WG4 (4 * HD)            // 2048 word gates

// padded LDS row strides (halfs): +8 halfs (4 dwords) kills 64-bank conflicts
#define CE_LD   72              // ce rows   (64  + 8)
#define H_LD    136             // h rows    (128 + 8)
#define X_LD    392             // xs rows   (384 + 8)
#define O_LD    520             // os rows   (512 + 8)
#define W_LD    520             // TDM-padded weight rows (512 + 8)

// ---- workspace layout (bytes, 256-aligned) ----
#define O_CWIH  0u                      // bf16 [CE][CG]      = 65536
#define O_CWHH  65536u                  // bf16 [CD][CG]      = 131072
#define O_CB    196608u                 // f32  [CG]          = 2048
#define O_WB    198656u                 // f32  [WG4]         = 8192
#define O_WOUT  206848u                 // bf16 [HD][TAGS]    = 49152
#define O_WWIH  256000u                 // bf16 [XDIM][WG4]   = 1572864
#define O_WWHH  1828864u                // bf16 [WG4][HD]     = 2097152
#define O_X     3926016u                // bf16 [NW][XDIM]    = 6291456
#define O_GX    10217472u               // f32  [NW][WG4]     = 67108864
#define O_OUTS  77326336u               // bf16 [NW][HD]      = 8388608

typedef __attribute__((ext_vector_type(16))) __bf16 v16bf;
typedef __attribute__((ext_vector_type(8)))  float  v8f;
typedef __attribute__((ext_vector_type(4)))  unsigned int u32x4;
typedef __attribute__((ext_vector_type(8)))  int i32x8;
typedef __attribute__((ext_vector_type(4)))  int i32x4;

union FragU { v16bf v; uint4 q[2]; unsigned short us[16]; };

__device__ __forceinline__ unsigned short f2b(float f) {
  unsigned int u = __float_as_uint(f);
  u += 0x7fffu + ((u >> 16) & 1u);          // round-to-nearest-even
  return (unsigned short)(u >> 16);
}
__device__ __forceinline__ float bl(unsigned int u) { return __uint_as_float(u << 16); }
__device__ __forceinline__ float bh(unsigned int u) { return __uint_as_float(u & 0xffff0000u); }
__device__ __forceinline__ float sigf(float x) { return 1.0f / (1.0f + __expf(-x)); }

__device__ __forceinline__ v8f wmma_bf16(v16bf a, v16bf b, v8f c) {
  return __builtin_amdgcn_wmma_f32_16x16x32_bf16(false, a, false, b, (short)0, c, false, false);
}

// A fragment (16x32 bf16): lane -> m = lane&15; elems 0..7: k = kBase + 8*(lane>>4)+i,
// elems 8..15: +16. Source row-major, row stride ld (halfs).
__device__ __forceinline__ v16bf load_a(const unsigned short* X, int ld, int kBase, int lane) {
  const unsigned short* p = X + (lane & 15) * ld + kBase + ((lane >> 4) << 3);
  FragU f;
  f.q[0] = *(const uint4*)(p);
  f.q[1] = *(const uint4*)(p + 16);
  return f.v;
}

// B fragment (32x16 bf16): lane -> n = nBase + (lane&15); elems 0..7: k = kBase + 8*(lane>>4)+i,
// elems 8..15: +16. Source = pre-transposed weights Wt[k][n], row stride ldn (halfs).
__device__ __forceinline__ v16bf load_b(const unsigned short* Wt, int ldn, int kBase, int nBase, int lane) {
  const unsigned short* p = Wt + (size_t)(kBase + ((lane >> 4) << 3)) * ldn + nBase + (lane & 15);
  FragU f;
  #pragma unroll
  for (int i = 0; i < 8; ++i) {
    f.us[i] = p[(size_t)i * ldn];
    f.us[8 + i] = p[(size_t)(16 + i) * ldn];
  }
  return f.v;
}

// ---- Tensor Data Mover: 2D bf16 tile (rows x 512) global -> LDS, with
//      4-dword padding after every 256 dwords (one 512-half row) so LDS row
//      stride becomes 1040 B (lane stride 4 banks for B-fragment ds_load_b128).
__device__ __forceinline__ void tdm_load_w(const unsigned short* g, unsigned lds_addr, int rows) {
  unsigned long long ga = (unsigned long long)(uintptr_t)g;
  u32x4 g0;
  g0.x = 1u;                                            // count=1, user D#
  g0.y = lds_addr;                                      // LDS byte address
  g0.z = (unsigned)(ga & 0xffffffffu);                  // global_addr[31:0]
  g0.w = (unsigned)((ga >> 32) & 0x1ffffffu) | (2u << 30); // addr[56:32] | type=2
  const unsigned dim0 = 512, tile0 = 512;
  const unsigned dim1 = (unsigned)rows, tile1 = (unsigned)rows;
  const unsigned long long s0 = 512, s1 = 512ull * rows;
  i32x8 g1;
  g1[0] = (int)((1u << 16) | (1u << 20) | (7u << 22) | (3u << 25)); // data_size=2B, pad on, every 256dw, 4dw
  g1[1] = (int)((dim0 & 0xffffu) << 16);                 // atomic_barrier=0 | dim0 lo
  g1[2] = (int)(((dim0 >> 16) & 0xffffu) | ((dim1 & 0xffffu) << 16));
  g1[3] = (int)(((dim1 >> 16) & 0xffffu) | ((tile0 & 0xffffu) << 16));
  g1[4] = (int)(tile1 & 0xffffu);                        // tile_dim1 | tile_dim2=0
  g1[5] = (int)(unsigned)(s0 & 0xffffffffu);             // dim0_stride lo
  g1[6] = (int)((unsigned)((s0 >> 32) & 0xffffu) | (unsigned)((s1 & 0xffffu) << 16));
  g1[7] = (int)(unsigned)((s1 >> 16) & 0xffffffffu);     // dim1_stride[47:16]
  i32x4 z4 = {0, 0, 0, 0};
#if defined(__clang_major__) && __clang_major__ >= 23
  i32x8 z8 = {0, 0, 0, 0, 0, 0, 0, 0};
  __builtin_amdgcn_tensor_load_to_lds(g0, g1, z4, z4, z8, 0);
#else
  __builtin_amdgcn_tensor_load_to_lds(g0, g1, z4, z4, 0);
#endif
}

// ---- async bf16 tile copy: global (contig rows of rowHalfs) -> LDS (row stride ldHalfs)
__device__ __forceinline__ void async_copy_row16(const unsigned short* gbase, unsigned gHalfOff,
                                                 unsigned ldsByteAddr) {
  asm volatile("global_load_async_to_lds_b128 %0, %1, %2"
               :: "v"(ldsByteAddr), "v"(gHalfOff * 2u), "s"(gbase) : "memory");
}
__device__ __forceinline__ void async_wait0() {
  asm volatile("s_wait_asynccnt 0" ::: "memory");
}

// ---------------- prep kernels ----------------
__global__ void cvt_transpose_kernel(const float* __restrict__ in, unsigned short* __restrict__ out,
                                     int R, int C) {  // in [R][C] -> out [C][R] bf16
  int i = blockIdx.x * blockDim.x + threadIdx.x;
  if (i < R * C) { int r = i / C, c = i - r * C; out[(size_t)c * R + r] = f2b(in[i]); }
}
__global__ void cvt_kernel(const float* __restrict__ in, unsigned short* __restrict__ out, int n) {
  int i = blockIdx.x * blockDim.x + threadIdx.x;
  if (i < n) out[i] = f2b(in[i]);
}
__global__ void add_bias_kernel(const float* __restrict__ a, const float* __restrict__ b,
                                float* __restrict__ out, int n) {
  int i = blockIdx.x * blockDim.x + threadIdx.x;
  if (i < n) out[i] = a[i] + b[i];
}
__global__ void word_embed_kernel(const int* __restrict__ word_ids, const float* __restrict__ w_emb,
                                  unsigned short* __restrict__ X) {
  int i = blockIdx.x * blockDim.x + threadIdx.x;
  if (i < NW * WE) {
    int s = i >> 8, d = i & (WE - 1);
    X[(size_t)s * XDIM + d] = f2b(w_emb[(size_t)word_ids[s] * WE + d]);
  }
}

// ---------------- char LSTM: 16 words/WG, 16 steps, WMMA recurrence ----------------
// dyn LDS layout (bytes):
//   wih  [64][520]  bf16 (TDM-padded) :  66560   (off 0)
//   whh  [128][520] bf16 (TDM-padded) : 133120   (off 66560)
//   ce   [16t][16m][72] bf16          :  36864   (off 199680)
//   h    [16m][136] bf16              :   4352   (off 236544)
//   c    [16m][128] f32               :   8192   (off 240896)
//   g    [16m][512] f32               :  32768   (off 249088)
//   cb   [512] f32                    :   2048   (off 281856)
//   len  [16] int                     :     64   (off 283904)  total 283968
#define CHAR_SMEM 283968u
__global__ __launch_bounds__(256) void char_lstm_kernel(
    const int* __restrict__ char_ids, const int* __restrict__ char_lens,
    const float* __restrict__ c_emb,
    const unsigned short* __restrict__ cWih_t, const unsigned short* __restrict__ cWhh_t,
    const float* __restrict__ cb, unsigned short* __restrict__ Xout) {
  extern __shared__ __align__(16) unsigned char smem[];
  unsigned short* wih_s = (unsigned short*)(smem);
  unsigned short* whh_s = (unsigned short*)(smem + 66560);
  unsigned short* ce_s  = (unsigned short*)(smem + 199680);
  unsigned short* h_s   = (unsigned short*)(smem + 236544);
  float* c_s   = (float*)(smem + 240896);
  float* g_s   = (float*)(smem + 249088);
  float* cb_s  = (float*)(smem + 281856);
  int*   len_s = (int*)(smem + 283904);

  const int tid = threadIdx.x;
  const int base = blockIdx.x * 16;

  // TDM: stage both weight matrices into LDS (wave 0 issues; overlaps gather below)
  if (tid < 32) {
    tdm_load_w(cWih_t, (unsigned)(uintptr_t)wih_s, CE);
    tdm_load_w(cWhh_t, (unsigned)(uintptr_t)whh_s, CD);
  }

  for (int e = tid; e < 16 * LC * CE; e += 256) {      // gather char embeddings -> bf16
    int m = e >> 10, rem = e & 1023, t = rem >> 6, d = rem & 63;
    int cid = char_ids[(base + m) * LC + t];
    ce_s[(t * 16 + m) * CE_LD + d] = f2b(c_emb[(size_t)cid * CE + d]);
  }
  for (int e = tid; e < 16 * H_LD; e += 256) h_s[e] = 0;
  for (int e = tid; e < 16 * CD; e += 256) c_s[e] = 0.0f;
  for (int e = tid; e < CG; e += 256) cb_s[e] = cb[e];
  if (tid < 16) len_s[tid] = char_lens[base + tid];

  if (tid < 32) __builtin_amdgcn_s_wait_tensorcnt(0);
  __syncthreads();

  const int wave = tid >> 5, lane = tid & 31;
  for (int t = 0; t < LC; ++t) {
    v16bf aX0 = load_a(ce_s + t * 16 * CE_LD, CE_LD, 0, lane);
    v16bf aX1 = load_a(ce_s + t * 16 * CE_LD, CE_LD, 32, lane);
    v16bf aH0 = load_a(h_s, H_LD, 0, lane);
    v16bf aH1 = load_a(h_s, H_LD, 32, lane);
    v16bf aH2 = load_a(h_s, H_LD, 64, lane);
    v16bf aH3 = load_a(h_s, H_LD, 96, lane);
    #pragma unroll
    for (int nt = 0; nt < 4; ++nt) {
      const int nBase = wave * 64 + nt * 16;
      v8f acc = {0.f, 0.f, 0.f, 0.f, 0.f, 0.f, 0.f, 0.f};
      acc = wmma_bf16(aX0, load_b(wih_s, W_LD, 0, nBase, lane), acc);
      acc = wmma_bf16(aX1, load_b(wih_s, W_LD, 32, nBase, lane), acc);
      acc = wmma_bf16(aH0, load_b(whh_s, W_LD, 0, nBase, lane), acc);
      acc = wmma_bf16(aH1, load_b(whh_s, W_LD, 32, nBase, lane), acc);
      acc = wmma_bf16(aH2, load_b(whh_s, W_LD, 64, nBase, lane), acc);
      acc = wmma_bf16(aH3, load_b(whh_s, W_LD, 96, nBase, lane), acc);
      const int n = nBase + (lane & 15);
      const int mh = (lane >> 4) << 3;
      #pragma unroll
      for (int v = 0; v < 8; ++v) g_s[(mh + v) * CG + n] = acc[v];
    }
    __syncthreads();
    for (int e = tid; e < 16 * CD; e += 256) {          // gate nonlinearity + ragged mask
      int m = e >> 7, j = e & 127;
      if (t < len_s[m]) {
        float gi = g_s[m * CG + j]            + cb_s[j];
        float gf = g_s[m * CG + CD + j]       + cb_s[CD + j];
        float gg = g_s[m * CG + 2 * CD + j]   + cb_s[2 * CD + j];
        float go = g_s[m * CG + 3 * CD + j]   + cb_s[3 * CD + j];
        float c2 = sigf(gf) * c_s[e] + sigf(gi) * tanhf(gg);
        float h2 = sigf(go) * tanhf(c2);
        c_s[e] = c2;
        h_s[m * H_LD + j] = f2b(h2);
      }
    }
    __syncthreads();
  }
  for (int e = tid; e < 16 * CD; e += 256) {            // ch -> X[:, 256:384]
    int m = e >> 7, j = e & 127;
    Xout[(size_t)(base + m) * XDIM + WE + j] = h_s[m * H_LD + j];
  }
}

// ---------------- word LSTM input GEMM: Gx = X @ wW_ih^T ----------------
__global__ __launch_bounds__(256) void input_gemm_kernel(
    const unsigned short* __restrict__ X, const unsigned short* __restrict__ wWih_t,
    float* __restrict__ Gx) {
  __shared__ __align__(16) unsigned short xs[16 * X_LD];
  const int tid = threadIdx.x;
  const int mbase = blockIdx.x * 16;
  // async copy 16 rows x 192 halfs (24 x 16B chunks per row) into padded LDS
  for (int c = tid; c < 16 * 24; c += 256) {
    int r = c / 24, col = c - r * 24;
    unsigned gOff = (unsigned)((mbase + r) * XDIM + col * 8);
    unsigned lAdr = (unsigned)(uintptr_t)(xs + r * X_LD + col * 8);
    async_copy_row16(X, gOff, lAdr);
  }
  async_wait0();
  __syncthreads();

  const int wave = tid >> 5, lane = tid & 31;
  const int nBase = blockIdx.y * 256 + wave * 32;
  v8f acc0 = {0.f, 0.f, 0.f, 0.f, 0.f, 0.f, 0.f, 0.f};
  v8f acc1 = {0.f, 0.f, 0.f, 0.f, 0.f, 0.f, 0.f, 0.f};
  #pragma unroll
  for (int kf = 0; kf < XDIM / 32; ++kf) {
    v16bf a = load_a(xs, X_LD, kf * 32, lane);
    acc0 = wmma_bf16(a, load_b(wWih_t, WG4, kf * 32, nBase, lane), acc0);
    acc1 = wmma_bf16(a, load_b(wWih_t, WG4, kf * 32, nBase + 16, lane), acc1);
  }
  const int mh = (lane >> 4) << 3, nl = lane & 15;
  #pragma unroll
  for (int v = 0; v < 8; ++v) {
    Gx[(size_t)(mbase + mh + v) * WG4 + nBase + nl]      = acc0[v];
    Gx[(size_t)(mbase + mh + v) * WG4 + nBase + 16 + nl] = acc1[v];
  }
}

// ---------------- word LSTM recurrence: single persistent WG, 8192 steps ----------------
__global__ __launch_bounds__(1024) void word_lstm_kernel(
    const unsigned short* __restrict__ wWhh, const float* __restrict__ wb,
    const float* __restrict__ Gx, unsigned short* __restrict__ outs) {
  __shared__ float hs[HD];
  __shared__ float gs[WG4];
  __shared__ float wbs[WG4];
  const int tid = threadIdx.x;
  for (int e = tid; e < WG4; e += 1024) wbs[e] = wb[e];
  if (tid < HD) hs[tid] = 0.0f;
  float creg = 0.0f;
  __syncthreads();

  for (int t = 0; t < NW; ++t) {
    __builtin_prefetch((const void*)(Gx + (size_t)(t + 1) * WG4 + tid), 0, 0);
    #pragma unroll
    for (int half = 0; half < 2; ++half) {
      const int g = tid + half * 1024;
      const uint4* wrow = (const uint4*)(wWhh + (size_t)g * HD);
      float acc = Gx[(size_t)t * WG4 + g] + wbs[g];
      #pragma unroll 8
      for (int kb = 0; kb < HD / 8; ++kb) {
        uint4 q = wrow[kb];
        const int k = kb * 8;
        acc = fmaf(bl(q.x), hs[k + 0], acc); acc = fmaf(bh(q.x), hs[k + 1], acc);
        acc = fmaf(bl(q.y), hs[k + 2], acc); acc = fmaf(bh(q.y), hs[k + 3], acc);
        acc = fmaf(bl(q.z), hs[k + 4], acc); acc = fmaf(bh(q.z), hs[k + 5], acc);
        acc = fmaf(bl(q.w), hs[k + 6], acc); acc = fmaf(bh(q.w), hs[k + 7], acc);
      }
      gs[g] = acc;
    }
    __syncthreads();
    if (tid < HD) {
      const int j = tid;
      float gi = gs[j], gf = gs[HD + j], gg = gs[2 * HD + j], go = gs[3 * HD + j];
      float c2 = sigf(gf) * creg + sigf(gi) * tanhf(gg);
      float h2 = sigf(go) * tanhf(c2);
      creg = c2;
      hs[j] = h2;
      outs[(size_t)t * HD + j] = f2b(h2);
    }
    __syncthreads();
  }
}

// ---------------- logits GEMM + fused log_softmax ----------------
__global__ __launch_bounds__(128) void logits_kernel(
    const unsigned short* __restrict__ outs, const unsigned short* __restrict__ Wout_t,
    const float* __restrict__ b_out, float* __restrict__ out) {
  __shared__ __align__(16) unsigned short os[16 * O_LD];
  __shared__ float lg[16 * TAGS];
  __shared__ float bo[TAGS];
  const int tid = threadIdx.x;
  const int mbase = blockIdx.x * 16;
  // async copy 16 rows x 512 halfs (64 x 16B chunks per row) into padded LDS
  for (int c = tid; c < 16 * 64; c += 128) {
    int r = c >> 6, col = c & 63;
    unsigned gOff = (unsigned)((mbase + r) * HD + col * 8);
    unsigned lAdr = (unsigned)(uintptr_t)(os + r * O_LD + col * 8);
    async_copy_row16(outs, gOff, lAdr);
  }
  if (tid < TAGS) bo[tid] = b_out[tid];
  async_wait0();
  __syncthreads();

  const int wave = tid >> 5, lane = tid & 31;
  if (wave < 3) {                         // 3 N-tiles of 16 = 48 tags; wave-uniform branch
    const int nBase = wave * 16;
    v8f acc = {0.f, 0.f, 0.f, 0.f, 0.f, 0.f, 0.f, 0.f};
    #pragma unroll
    for (int kf = 0; kf < HD / 32; ++kf) {
      v16bf a = load_a(os, O_LD, kf * 32, lane);
      acc = wmma_bf16(a, load_b(Wout_t, TAGS, kf * 32, nBase, lane), acc);
    }
    const int mh = (lane >> 4) << 3, nl = lane & 15;
    #pragma unroll
    for (int v = 0; v < 8; ++v) lg[(mh + v) * TAGS + nBase + nl] = acc[v] + bo[nBase + nl];
  }
  __syncthreads();
  if (tid < 16) {
    const int r = tid;
    float mx = -1e30f;
    for (int n = 0; n < TAGS; ++n) mx = fmaxf(mx, lg[r * TAGS + n]);
    float s = 0.0f;
    for (int n = 0; n < TAGS; ++n) s += __expf(lg[r * TAGS + n] - mx);
    const float lse = mx + __logf(s);
    for (int n = 0; n < TAGS; ++n) out[(size_t)(mbase + r) * TAGS + n] = lg[r * TAGS + n] - lse;
  }
}

// ---------------------------------------------------------------------------
extern "C" void kernel_launch(void* const* d_in, const int* in_sizes, int n_in,
                              void* d_out, int out_size, void* d_ws, size_t ws_size,
                              hipStream_t stream) {
  (void)in_sizes; (void)n_in; (void)out_size; (void)ws_size;
  const int*   word_ids  = (const int*)d_in[0];
  const int*   char_ids  = (const int*)d_in[1];
  const int*   char_lens = (const int*)d_in[2];
  const float* w_emb = (const float*)d_in[3];
  const float* c_emb = (const float*)d_in[4];
  const float* cW_ih = (const float*)d_in[5];
  const float* cW_hh = (const float*)d_in[6];
  const float* cb_ih = (const float*)d_in[7];
  const float* cb_hh = (const float*)d_in[8];
  const float* wW_ih = (const float*)d_in[9];
  const float* wW_hh = (const float*)d_in[10];
  const float* wb_ih = (const float*)d_in[11];
  const float* wb_hh = (const float*)d_in[12];
  const float* W_out = (const float*)d_in[13];
  const float* b_out = (const float*)d_in[14];

  unsigned char* ws = (unsigned char*)d_ws;
  unsigned short* cWih_t = (unsigned short*)(ws + O_CWIH);
  unsigned short* cWhh_t = (unsigned short*)(ws + O_CWHH);
  float*          cb     = (float*)(ws + O_CB);
  float*          wb     = (float*)(ws + O_WB);
  unsigned short* Wout_t = (unsigned short*)(ws + O_WOUT);
  unsigned short* wWih_t = (unsigned short*)(ws + O_WWIH);
  unsigned short* wWhh   = (unsigned short*)(ws + O_WWHH);
  unsigned short* X      = (unsigned short*)(ws + O_X);
  float*          Gx     = (float*)(ws + O_GX);
  unsigned short* outs   = (unsigned short*)(ws + O_OUTS);

  // --- prep: bf16 conversion / transposes / bias sums ---
  cvt_transpose_kernel<<<(CG * CE + 255) / 256, 256, 0, stream>>>(cW_ih, cWih_t, CG, CE);
  cvt_transpose_kernel<<<(CG * CD + 255) / 256, 256, 0, stream>>>(cW_hh, cWhh_t, CG, CD);
  cvt_transpose_kernel<<<(WG4 * XDIM + 255) / 256, 256, 0, stream>>>(wW_ih, wWih_t, WG4, XDIM);
  cvt_transpose_kernel<<<(TAGS * HD + 255) / 256, 256, 0, stream>>>(W_out, Wout_t, TAGS, HD);
  cvt_kernel<<<(WG4 * HD + 255) / 256, 256, 0, stream>>>(wW_hh, wWhh, WG4 * HD);
  add_bias_kernel<<<2, 256, 0, stream>>>(cb_ih, cb_hh, cb, CG);
  add_bias_kernel<<<8, 256, 0, stream>>>(wb_ih, wb_hh, wb, WG4);
  word_embed_kernel<<<(NW * WE + 255) / 256, 256, 0, stream>>>(word_ids, w_emb, X);

  // --- char LSTM (WMMA recurrence; weights staged via Tensor Data Mover) ---
  char_lstm_kernel<<<NW / 16, 256, CHAR_SMEM, stream>>>(char_ids, char_lens, c_emb,
                                                        cWih_t, cWhh_t, cb, X);

  // --- word LSTM input GEMM (WMMA; async LDS staging) ---
  input_gemm_kernel<<<dim3(NW / 16, WG4 / 256), 256, 0, stream>>>(X, wWih_t, Gx);

  // --- word LSTM recurrence (sequential, single WG) ---
  word_lstm_kernel<<<1, 1024, 0, stream>>>(wWhh, wb, Gx, outs);

  // --- logits + log_softmax (WMMA; async LDS staging) ---
  logits_kernel<<<NW / 16, 128, 0, stream>>>(outs, Wout_t, b_out, (float*)d_out);
}
